// SPNet_17197049053424
// MI455X (gfx1250) — matrix operand order repaired and checked by reading, compiled, not run on hardware
//
#include <hip/hip_runtime.h>

typedef unsigned short ushort_t;
typedef __attribute__((ext_vector_type(16))) __bf16 v16bf;
typedef __attribute__((ext_vector_type(8)))  __bf16 v8bf;
typedef __attribute__((ext_vector_type(8)))  float  v8f;

#define BGRID 2
#define DGRID 128
#define HGRID 128
#define WGRID 128
#define EPSBN 1e-5f
#define CONVBLK 512   // 16 waves per block, each wave owns a 32x64 output tile

__device__ __forceinline__ ushort_t f2bf(float x) {
    unsigned u = __builtin_bit_cast(unsigned, x);
    unsigned r = (u + 0x7FFFu + ((u >> 16) & 1u)) >> 16;   // round-to-nearest-even
    return (ushort_t)r;
}

// ---------------- rulebook construction ----------------
__global__ void init_table(int* __restrict__ table, int n) {
    int i = blockIdx.x * blockDim.x + threadIdx.x;
    if (i < n) table[i] = -1;
}

__global__ void scatter_table(const int* __restrict__ coords, int* __restrict__ table, int nvox) {
    int i = blockIdx.x * blockDim.x + threadIdx.x;
    if (i >= nvox) return;
    int b = coords[i * 4 + 0], z = coords[i * 4 + 1];
    int y = coords[i * 4 + 2], x = coords[i * 4 + 3];
    int flat = ((b * DGRID + z) * HGRID + y) * WGRID + x;
    table[flat] = i;
}

__global__ void build_nbr(const int* __restrict__ coords, const int* __restrict__ table,
                          int* __restrict__ nbr, int nvox) {
    int i = blockIdx.x * blockDim.x + threadIdx.x;
    if (i >= nvox) return;
    int b = coords[i * 4 + 0], z = coords[i * 4 + 1];
    int y = coords[i * 4 + 2], x = coords[i * 4 + 3];
#pragma unroll
    for (int k = 0; k < 27; ++k) {
        int dz = k / 9 - 1, dy = (k / 3) % 3 - 1, dx = k % 3 - 1;
        int nz = z + dz, ny = y + dy, nx = x + dx;
        int val = -1;
        if (nz >= 0 && nz < DGRID && ny >= 0 && ny < HGRID && nx >= 0 && nx < WGRID) {
            int flat = ((b * DGRID + nz) * HGRID + ny) * WGRID + nx;
            val = table[flat];
        }
        nbr[(size_t)k * nvox + i] = val;
    }
}

// ---------------- weight prep: [k][ci][co] f32 -> [k][co][ci] bf16 ----------------
__global__ void transpose_w(const float* __restrict__ w, ushort_t* __restrict__ wt) {
    int i = blockIdx.x * blockDim.x + threadIdx.x;   // 27*64*64
    if (i >= 27 * 4096) return;
    int k = i >> 12, r = i & 4095, ci = r >> 6, co = r & 63;
    wt[k * 4096 + co * 64 + ci] = f2bf(w[i]);
}

__global__ void zero_rows(ushort_t* __restrict__ x1, ushort_t* __restrict__ x2, int nvox) {
    int t = threadIdx.x;
    if (t < 64) x1[(size_t)nvox * 64 + t] = 0;
    else        x2[(size_t)nvox * 64 + (t - 64)] = 0;
}

// ---------------- layer 1: cin=1 (pure VALU, memory-light) ----------------
__global__ __launch_bounds__(256)
void conv1_bn(const float* __restrict__ feats, const float* __restrict__ w1,
              const float* __restrict__ b1, const float* __restrict__ g1,
              const float* __restrict__ be1, const float* __restrict__ m1,
              const float* __restrict__ v1, const int* __restrict__ nbr,
              ushort_t* __restrict__ x1, int nvox) {
    int c = threadIdx.x & 63;
    int q = threadIdx.x >> 6;
    int i = blockIdx.x * 4 + q;
    if (i >= nvox) return;
    float acc = b1[c];
#pragma unroll
    for (int k = 0; k < 27; ++k) {
        int idx = nbr[(size_t)k * nvox + i];
        float f = (idx >= 0) ? feats[idx] : 0.f;
        acc = fmaf(f, w1[k * 64 + c], acc);
    }
    float sc = g1[c] * rsqrtf(v1[c] + EPSBN);
    float y = (acc - m1[c]) * sc + be1[c];
    y = y > 0.f ? y : 0.f;
    x1[(size_t)i * 64 + c] = f2bf(y);
}

// ---------------- layers 2/3: 64->64 WMMA gather-GEMM, weights in LDS ----------------
// Each wave: 32 voxels x 64 channels = 8 C tiles; every LDS B fragment feeds 2 WMMAs.
template <bool WRITE_F32>
__global__ __launch_bounds__(CONVBLK)
void subm_conv64(const ushort_t* __restrict__ xin,   // (nvox+pad) x 64 bf16, row nvox = zeros
                 const ushort_t* __restrict__ wt,    // 27 x 64(co) x 64(ci) bf16
                 const int* __restrict__ nbr,        // 27 x nvox
                 const float* __restrict__ bias,
                 const float* __restrict__ gg, const float* __restrict__ be,
                 const float* __restrict__ mm, const float* __restrict__ vv,
                 ushort_t* __restrict__ obf, float* __restrict__ of32, int nvox) {
    extern __shared__ ushort_t smem[];               // 27*4096 bf16 = 221184 B
    {
        const uint4* s = (const uint4*)wt;
        uint4* d = (uint4*)smem;
        for (int i = threadIdx.x; i < 13824; i += CONVBLK) d[i] = s[i];
    }
    __syncthreads();

    const int lane = threadIdx.x & 31;
    const int wv   = threadIdx.x >> 5;
    const int tile = blockIdx.x * (CONVBLK / 32) + wv;
    const int v0   = tile * 32;
    if (v0 >= nvox) return;

    const int row   = lane & 15;      // A: M row / B,C: N column
    const int hi    = lane >> 4;      // lane half selects K sub-range
    const int khalf = hi * 8;

    // hoist BN affine params so the epilogue has no loads
    float scv[4], shv[4], bv[4];
#pragma unroll
    for (int t = 0; t < 4; ++t) {
        int ch = t * 16 + row;
        bv[t]  = bias[ch];
        float sc = gg[ch] * rsqrtf(vv[ch] + EPSBN);
        scv[t] = sc;
        shv[t] = be[ch] - mm[ch] * sc;
    }

    v8f acc0[4], acc1[4];             // M rows 0..15 / 16..31
#pragma unroll
    for (int t = 0; t < 4; ++t)
#pragma unroll
        for (int r = 0; r < 8; ++r) { acc0[t][r] = bv[t]; acc1[t][r] = bv[t]; }

#pragma unroll
    for (int k = 0; k < 27; ++k) {
        const int* nb = nbr + (size_t)k * nvox + v0;
        int ia = nb[row];       ia = (ia < 0) ? nvox : ia;      // inactive -> zero row
        int ib = nb[16 + row];  ib = (ib < 0) ? nvox : ib;
        const ushort_t* ra = xin + (size_t)ia * 64;
        const ushort_t* rb = xin + (size_t)ib * 64;
        // A fragments, 16-bit 16x32 layout: lane half hi covers K = khalf..+7 and 16+khalf..+7
        v8bf a00 = *(const v8bf*)(ra + khalf);        // M-half 0, cin  0..31
        v8bf a01 = *(const v8bf*)(ra + khalf + 16);
        v8bf a10 = *(const v8bf*)(ra + khalf + 32);   // M-half 0, cin 32..63
        v8bf a11 = *(const v8bf*)(ra + khalf + 48);
        v8bf c00 = *(const v8bf*)(rb + khalf);        // M-half 1
        v8bf c01 = *(const v8bf*)(rb + khalf + 16);
        v8bf c10 = *(const v8bf*)(rb + khalf + 32);
        v8bf c11 = *(const v8bf*)(rb + khalf + 48);
        v16bf Aa0, Aa1, Ab0, Ab1;
#pragma unroll
        for (int e = 0; e < 8; ++e) {
            Aa0[e] = a00[e]; Aa0[8 + e] = a01[e];
            Aa1[e] = a10[e]; Aa1[8 + e] = a11[e];
            Ab0[e] = c00[e]; Ab0[8 + e] = c01[e];
            Ab1[e] = c10[e]; Ab1[8 + e] = c11[e];
        }
        const ushort_t* wk = smem + k * 4096;
#pragma unroll
        for (int t = 0; t < 4; ++t) {
            // B fragment: lane holds column N=row of W^T, cin khalf..+7 and 16+khalf..+7
            const ushort_t* col = wk + (t * 16 + row) * 64;
            v8bf b00 = *(const v8bf*)(col + khalf);
            v8bf b01 = *(const v8bf*)(col + khalf + 16);
            v8bf b10 = *(const v8bf*)(col + khalf + 32);
            v8bf b11 = *(const v8bf*)(col + khalf + 48);
            v16bf B0 = __builtin_shufflevector(b00, b01, 0, 1, 2, 3, 4, 5, 6, 7, 8, 9, 10, 11, 12, 13, 14, 15);        // cin  0..31
            v16bf B1 = __builtin_shufflevector(b10, b11, 0, 1, 2, 3, 4, 5, 6, 7, 8, 9, 10, 11, 12, 13, 14, 15);   // cin 32..63
            acc0[t] = __builtin_amdgcn_wmma_f32_16x16x32_bf16(
                false, Aa0, false, B0, (short)0, acc0[t], false, false);
            acc1[t] = __builtin_amdgcn_wmma_f32_16x16x32_bf16(
                false, Ab0, false, B0, (short)0, acc1[t], false, false);
            acc0[t] = __builtin_amdgcn_wmma_f32_16x16x32_bf16(
                false, Aa1, false, B1, (short)0, acc0[t], false, false);
            acc1[t] = __builtin_amdgcn_wmma_f32_16x16x32_bf16(
                false, Ab1, false, B1, (short)0, acc1[t], false, false);
        }
    }

    // fused BN + ReLU epilogue; C layout: ch = t*16 + (lane&15), vox row = r + 8*hi
#pragma unroll
    for (int h = 0; h < 2; ++h) {
#pragma unroll
        for (int t = 0; t < 4; ++t) {
            v8f* a = h ? acc1 : acc0;
#pragma unroll
            for (int r = 0; r < 8; ++r) {
                float y = a[t][r] * scv[t] + shv[t];
                y = y > 0.f ? y : 0.f;
                size_t o = (size_t)(v0 + h * 16 + r + hi * 8) * 64 + (t * 16 + row);
                if (WRITE_F32) of32[o] = y;
                else           obf[o] = f2bf(y);
            }
        }
    }
}

extern "C" void kernel_launch(void* const* d_in, const int* in_sizes, int n_in,
                              void* d_out, int out_size, void* d_ws, size_t ws_size,
                              hipStream_t stream) {
    (void)n_in; (void)out_size; (void)ws_size;
    const int nvox = in_sizes[0];                 // feats is (N,1)

    const float* feats = (const float*)d_in[0];
    const int*   coords = (const int*)d_in[1];
    const float* w1 = (const float*)d_in[2];
    const float* b1 = (const float*)d_in[3];
    const float* g1 = (const float*)d_in[4];
    const float* be1 = (const float*)d_in[5];
    const float* m1 = (const float*)d_in[6];
    const float* v1 = (const float*)d_in[7];
    const float* w2 = (const float*)d_in[8];
    const float* b2 = (const float*)d_in[9];
    const float* g2 = (const float*)d_in[10];
    const float* be2 = (const float*)d_in[11];
    const float* m2 = (const float*)d_in[12];
    const float* v2 = (const float*)d_in[13];
    const float* w3 = (const float*)d_in[14];
    const float* b3 = (const float*)d_in[15];
    const float* g3 = (const float*)d_in[16];
    const float* be3 = (const float*)d_in[17];
    const float* m3 = (const float*)d_in[18];
    const float* v3 = (const float*)d_in[19];
    float* out = (float*)d_out;

    char* ws = (char*)d_ws;
    auto a256 = [](size_t x) { return (x + 255) & ~(size_t)255; };
    size_t off = 0;
    const size_t tabN = (size_t)BGRID * DGRID * HGRID * WGRID;
    int* table = (int*)(ws + off);       off = a256(off + tabN * 4);
    int* nbr   = (int*)(ws + off);       off = a256(off + (size_t)27 * nvox * 4);
    ushort_t* x1  = (ushort_t*)(ws + off); off = a256(off + (size_t)(nvox + 32) * 64 * 2);
    ushort_t* x2  = (ushort_t*)(ws + off); off = a256(off + (size_t)(nvox + 32) * 64 * 2);
    ushort_t* w2t = (ushort_t*)(ws + off); off = a256(off + (size_t)27 * 4096 * 2);
    ushort_t* w3t = (ushort_t*)(ws + off); off = a256(off + (size_t)27 * 4096 * 2);

    init_table<<<(int)((tabN + 255) / 256), 256, 0, stream>>>(table, (int)tabN);
    scatter_table<<<(nvox + 255) / 256, 256, 0, stream>>>(coords, table, nvox);
    build_nbr<<<(nvox + 255) / 256, 256, 0, stream>>>(coords, table, nbr, nvox);
    transpose_w<<<(27 * 4096 + 255) / 256, 256, 0, stream>>>(w2, w2t);
    transpose_w<<<(27 * 4096 + 255) / 256, 256, 0, stream>>>(w3, w3t);
    zero_rows<<<1, 128, 0, stream>>>(x1, x2, nvox);

    conv1_bn<<<(nvox + 3) / 4, 256, 0, stream>>>(feats, w1, b1, g1, be1, m1, v1, nbr, x1, nvox);

    int tiles = (nvox + 31) / 32;                 // 32-voxel wave tiles
    int tilesPerBlk = CONVBLK / 32;
    int blocks = (tiles + tilesPerBlk - 1) / tilesPerBlk;
    size_t shmem = (size_t)27 * 4096 * 2;         // 221184 B of the 320 KB WGP LDS
    subm_conv64<false><<<blocks, CONVBLK, shmem, stream>>>(x1, w2t, nbr, b2, g2, be2, m2, v2,
                                                           x2, nullptr, nvox);
    subm_conv64<true><<<blocks, CONVBLK, shmem, stream>>>(x2, w3t, nbr, b3, g3, be3, m3, v3,
                                                          nullptr, out, nvox);
}
